// Pe_global_based_attention_variation_33122787787030
// MI455X (gfx1250) — matrix operand order repaired and checked by reading, not compiled, not run on hardware
//
#include <hip/hip_runtime.h>
#include <math.h>

#define BB 4
#define HH 8
#define NN 2048
#define DD 64
#define BM 64            // query rows per workgroup
#define BN 64            // key rows per tile
#define KSTR 136         // Kaug LDS stride (bf16 elems), 272B row: 16B aligned, bank-skewed
#define VSTR 72          // V^T LDS stride
#define PSTR 72          // P LDS stride
#define NWAVES 4

typedef __attribute__((ext_vector_type(16))) __bf16 v16bf;
typedef __attribute__((ext_vector_type(8)))  float  v8f;

union Frag {
  uint4 q[2];
  v16bf v;
  unsigned u32[8];
};

__device__ __forceinline__ unsigned short f2bf(float f) {
  unsigned u = __builtin_bit_cast(unsigned, f);
  u += 0x7FFFu + ((u >> 16) & 1u);          // round-to-nearest-even
  return (unsigned short)(u >> 16);
}
__device__ __forceinline__ unsigned pk2(float a, float b) {
  return (unsigned)f2bf(a) | ((unsigned)f2bf(b) << 16);
}

__global__ __launch_bounds__(128)
void pe_attn_kernel(const float* __restrict__ q, const float* __restrict__ k,
                    const float* __restrict__ v, const float* __restrict__ xyz,
                    const float* __restrict__ W1, const float* __restrict__ W2,
                    const float* __restrict__ W3, float* __restrict__ out) {
  __shared__ __attribute__((aligned(16))) unsigned short sK[BN * KSTR];   // Kaug bf16
  __shared__ __attribute__((aligned(16))) unsigned short sV[DD * VSTR];   // V^T bf16
  __shared__ __attribute__((aligned(16))) unsigned short sP[NWAVES * 16 * PSTR];
  __shared__ float4 sWk4[128];   // rows 0..63 = W1[h], 64..127 = W3[h]
  __shared__ float4 sWv4[64];    // W2[h]

  const int tid  = threadIdx.x;
  const int lane = tid & 31;
  const int wave = tid >> 5;
  const int m    = lane & 15;    // within-tile row (query m / A-matrix row)
  const int hf   = lane >> 4;    // lane half

  const int bh = blockIdx.y;
  const int b  = bh / HH, h = bh % HH;
  const int qbase = blockIdx.x * BM;

  // --- stage per-head PE weights into LDS (padded float4 rows) ---
  if (tid < 128) {
    const float* src = (tid < 64) ? (W1 + (size_t)(h * DD + tid) * 3)
                                  : (W3 + (size_t)(h * DD + tid - 64) * 3);
    sWk4[tid] = make_float4(src[0], src[1], src[2], 0.0f);
    if (tid < 64) {
      const float* s2 = W2 + (size_t)(h * DD + tid) * 3;
      sWv4[tid] = make_float4(s2[0], s2[1], s2[2], 0.0f);
    }
  }

  // --- Qaug B-fragments (held in registers whole kernel), scaled by 1/sqrt(D) ---
  // B layout (32x16): lane = N col (query m), half hf, vgpr i packs K = 2i + 16*hf + 32*chunk
  v16bf qb[4];
  {
    const int qrow = qbase + wave * 16 + m;
    const float* qrowp = q + ((size_t)(b * HH + h) * NN + qrow) * DD;
    const float* krowp = k + ((size_t)(b * HH + h) * NN + qrow) * DD;
#pragma unroll
    for (int c = 0; c < 4; c++) {
      const float* src = (c < 2 ? qrowp : krowp) + (c & 1) * 32 + hf * 16;
      Frag f;
#pragma unroll
      for (int i = 0; i < 8; i++)
        f.u32[i] = pk2(src[2 * i] * 0.125f, src[2 * i + 1] * 0.125f);
      qb[c] = f.v;
    }
  }

  v8f zero = {0.f, 0.f, 0.f, 0.f, 0.f, 0.f, 0.f, 0.f};
  v8f oacc[4] = {zero, zero, zero, zero};   // O^T accumulator: 4 d-subtiles x (16x16 f32)
  float run_m = -3.0e38f;
  float run_l = 0.0f;

  const float* kbase = k + (size_t)(b * HH + h) * NN * DD;
  const float* vbase = v + (size_t)(b * HH + h) * NN * DD;
  const float* xb    = xyz + (size_t)b * 3 * NN;

  for (int t0 = 0; t0 < NN; t0 += BN) {
    __syncthreads();   // protect LDS rewrite; covers weight staging on first iter

    // ---------- cooperative tile build: Kaug (k + q_pe^T | k_pe^T), V^T (v + v_pe^T) ----------
    {
      const int kn  = tid >> 1;          // 0..63 key row in tile
      const int s   = tid & 1;           // which 32-wide column half
      const int kng = t0 + kn;
      const float x0 = xb[kng];
      const float x1 = xb[NN + kng];
      const float x2 = xb[2 * NN + kng];
      const float* krow = kbase + (size_t)kng * DD + s * 32;
      const float* vrow = vbase + (size_t)kng * DD + s * 32;
      if (t0 + BN < NN) {                // speculative prefetch of next tile
        __builtin_prefetch(krow + BN * DD, 0, 1);
        __builtin_prefetch(vrow + BN * DD, 0, 1);
      }
#pragma unroll
      for (int j = 0; j < 16; j++) {
        const int c = s * 32 + 2 * j;    // Kaug column pair (0..63)
        float2 kk = ((const float2*)krow)[j];
        float2 vv = ((const float2*)vrow)[j];
        float4 w0 = sWk4[c];
        float4 w1 = sWk4[c + 1];
        float a0 = kk.x + w0.x * x0 + w0.y * x1 + w0.z * x2;   // k + q_pe
        float a1 = kk.y + w1.x * x0 + w1.y * x1 + w1.z * x2;
        *(unsigned*)&sK[kn * KSTR + c] = pk2(a0, a1);
        float4 w2 = sWk4[64 + c];
        float4 w3 = sWk4[64 + c + 1];
        float e0 = w2.x * x0 + w2.y * x1 + w2.z * x2;           // k_pe
        float e1 = w3.x * x0 + w3.y * x1 + w3.z * x2;
        *(unsigned*)&sK[kn * KSTR + 64 + c] = pk2(e0, e1);
        float4 u0 = sWv4[c];
        float4 u1 = sWv4[c + 1];
        sV[c * VSTR + kn]       = f2bf(vv.x + u0.x * x0 + u0.y * x1 + u0.z * x2);
        sV[(c + 1) * VSTR + kn] = f2bf(vv.y + u1.x * x0 + u1.y * x1 + u1.z * x2);
      }
    }
    __syncthreads();

    // ---------- S^T = Ktile(16x128 per subtile) x Qaug^T : 4 subtiles x 4 K-chunks ----------
    // A layout (16x32): lane m = K-tile row; vgprs 0-7 hold K=16*hf+[0..15]
    float st[4][8];
#pragma unroll
    for (int j = 0; j < 4; j++) {
      v8f acc = zero;
      const unsigned short* krow = &sK[(16 * j + m) * KSTR];
#pragma unroll
      for (int c = 0; c < 4; c++) {
        Frag a;
        a.q[0] = *(const uint4*)(krow + 32 * c + 16 * hf);
        a.q[1] = *(const uint4*)(krow + 32 * c + 16 * hf + 8);
        acc = __builtin_amdgcn_wmma_f32_16x16x32_bf16(false, a.v, false, qb[c],
                                                      (short)0, acc, false, false);
      }
#pragma unroll
      for (int r = 0; r < 8; r++) st[j][r] = acc[r];
    }

    // ---------- online softmax (each lane owns one query row m; halves split key range) ----------
    float tmax = st[0][0];
#pragma unroll
    for (int j = 0; j < 4; j++)
#pragma unroll
      for (int r = 0; r < 8; r++) tmax = fmaxf(tmax, st[j][r]);
    tmax = fmaxf(tmax, __shfl_xor(tmax, 16, 32));
    const float nm = fmaxf(run_m, tmax);
    const float sf = __expf(run_m - nm);
    float ts = 0.0f;
#pragma unroll
    for (int j = 0; j < 4; j++)
#pragma unroll
      for (int r = 0; r < 8; r++) {
        float p = __expf(st[j][r] - nm);
        st[j][r] = p;
        ts += p;
      }
    ts += __shfl_xor(ts, 16, 32);
    run_l = run_l * sf + ts;
    run_m = nm;
#pragma unroll
    for (int jd = 0; jd < 4; jd++) oacc[jd] *= sf;

    // ---------- P round-trip through per-wave LDS (C/D-layout rows are contiguous here) ----------
    unsigned short* prow = &sP[(wave * 16 + m) * PSTR];
#pragma unroll
    for (int j = 0; j < 4; j++) {
      Frag pf;
#pragma unroll
      for (int r = 0; r < 4; r++) pf.u32[r] = pk2(st[j][2 * r], st[j][2 * r + 1]);
      *(uint4*)(prow + 16 * j + 8 * hf) = pf.q[0];
    }

    // ---------- O^T += V^T x P^T : 4 d-subtiles x 2 kn-chunks ----------
#pragma unroll
    for (int c = 0; c < 2; c++) {
      Frag pb;   // B operand: lane = query m, vgpr i packs kn = 2i + 16*hf + 32*c
      pb.q[0] = *(const uint4*)(prow + 32 * c + 16 * hf);
      pb.q[1] = *(const uint4*)(prow + 32 * c + 16 * hf + 8);
#pragma unroll
      for (int jd = 0; jd < 4; jd++) {
        const unsigned short* vr = &sV[(16 * jd + m) * VSTR];
        Frag a;
        a.q[0] = *(const uint4*)(vr + 32 * c + 16 * hf);
        a.q[1] = *(const uint4*)(vr + 32 * c + 16 * hf + 8);
        oacc[jd] = __builtin_amdgcn_wmma_f32_16x16x32_bf16(false, a.v, false, pb.v,
                                                           (short)0, oacc[jd], false, false);
      }
    }
  }

  // ---------- normalize + store: out[b, n, h, d] ----------
  const float inv = 1.0f / run_l;
  const int n = qbase + wave * 16 + m;
  float* orow = out + ((size_t)(b * NN + n) * HH + h) * DD;
#pragma unroll
  for (int jd = 0; jd < 4; jd++) {
    const int d0 = 16 * jd + 8 * hf;     // O^T: vgpr r -> d = 16*jd + 8*hf + r
    float4 o0 = {oacc[jd][0] * inv, oacc[jd][1] * inv, oacc[jd][2] * inv, oacc[jd][3] * inv};
    float4 o1 = {oacc[jd][4] * inv, oacc[jd][5] * inv, oacc[jd][6] * inv, oacc[jd][7] * inv};
    *(float4*)(orow + d0)     = o0;
    *(float4*)(orow + d0 + 4) = o1;
  }
}

extern "C" void kernel_launch(void* const* d_in, const int* in_sizes, int n_in,
                              void* d_out, int out_size, void* d_ws, size_t ws_size,
                              hipStream_t stream) {
  const float* q   = (const float*)d_in[0];
  const float* k   = (const float*)d_in[1];
  const float* v   = (const float*)d_in[2];
  const float* xyz = (const float*)d_in[3];
  const float* W1  = (const float*)d_in[4];
  const float* W2  = (const float*)d_in[5];
  const float* W3  = (const float*)d_in[6];
  dim3 grid(NN / BM, BB * HH);
  pe_attn_kernel<<<grid, 128, 0, stream>>>(q, k, v, xyz, W1, W2, W3, (float*)d_out);
}
